// VQVAE_20684562497707
// MI455X (gfx1250) — compile-verified, matrix operand order checked by reading, not run on hardware
//
#include <hip/hip_runtime.h>
#include <hip/hip_bf16.h>

// ---------------------------------------------------------------------------
// Types for CDNA5 WMMA (gfx1250, wave32)
// ---------------------------------------------------------------------------
typedef __bf16 bf16_t;
typedef __attribute__((ext_vector_type(16))) __bf16 v16bf;
typedef __attribute__((ext_vector_type(8)))  __bf16 v8bf;
typedef __attribute__((ext_vector_type(8)))  float  v8f;

// 16x32 bf16 A-fragment row slice for this lane.
// Lane L holds row M = L%16; h = L/16 selects K halves:
//   elements e=0..7  -> K = h*8 + e
//   elements e=8..15 -> K = 16 + h*8 + (e-8)
__device__ inline v16bf load_a_frag(const bf16_t* __restrict__ row, int kbase, int h) {
    const v8bf lo = *(const v8bf*)(row + kbase + h * 8);
    const v8bf hi = *(const v8bf*)(row + kbase + h * 8 + 16);
    return __builtin_shufflevector(lo, hi, 0, 1, 2, 3, 4, 5, 6, 7,
                                           8, 9, 10, 11, 12, 13, 14, 15);
}

__device__ inline int clampi(int v, int lo, int hi) {
    return v < lo ? lo : (v > hi ? hi : v);
}

// ---------------------------------------------------------------------------
// Weight repack kernels (f32 -> padded bf16 GEMM matrices)
// ---------------------------------------------------------------------------
__global__ void vq_pack_conv_w(const float* __restrict__ src, bf16_t* __restrict__ dst,
                               int Cout, int CoutPad, int Kreal, int Kpad) {
    long i = (long)blockIdx.x * blockDim.x + threadIdx.x;
    long tot = (long)CoutPad * Kpad;
    if (i >= tot) return;
    int m = (int)(i / Kpad);
    int k = (int)(i % Kpad);
    float v = (m < Cout && k < Kreal) ? src[(long)m * Kreal + k] : 0.0f;
    dst[i] = (bf16_t)v;
}

// ConvTranspose weight [Cin][Cout][4][4] -> 4 parity classes [cls][CoutPad][Cin*4].
__global__ void vq_pack_convt_w(const float* __restrict__ src, bf16_t* __restrict__ dst,
                                int Cin, int Cout, int CoutPad) {
    int Kpad = Cin * 4;
    long i = (long)blockIdx.x * blockDim.x + threadIdx.x;
    long tot = 4L * CoutPad * Kpad;
    if (i >= tot) return;
    int cls = (int)(i / ((long)CoutPad * Kpad));
    long r = i % ((long)CoutPad * Kpad);
    int m = (int)(r / Kpad);
    int k = (int)(r % Kpad);
    int py = cls >> 1, px = cls & 1;
    int cin = k >> 2, t = k & 3, ty = t >> 1, tx = t & 1;
    int ky = 1 - py + 2 * ty;
    int kx = 1 - px + 2 * tx;
    float v = (m < Cout) ? src[(((long)cin * Cout + m) * 4 + ky) * 4 + kx] : 0.0f;
    dst[i] = (bf16_t)v;
}

__global__ void vq_zero_scalar(float* p) { if (threadIdx.x == 0 && blockIdx.x == 0) *p = 0.0f; }

// ---------------------------------------------------------------------------
// LDS-staged implicit-GEMM conv with bf16 WMMA.
// One workgroup (4 waves) owns ONE 16-position tile; the 4 waves cover
// different Cout ranges (4 * MT * 16 rows per grid.y slice) and share a
// cooperatively staged B tile in LDS (16 pos x 32 k bf16 per K-step).
// All geometry is compile-time (powers of two) -> decode is shifts/masks.
// Grid covers COUT exactly -> no M guard in the epilogue.
// ACT: 1 = relu -> bf16 out, 0 = linear -> f32 out.
// ---------------------------------------------------------------------------
template <typename InT, int ACT, int MT, int KK, int STRIDE, int PAD, int KREAL, int KPAD,
          int CIN, int HIN, int WIN, int COUT, int HO, int WO>
__global__ __launch_bounds__(128)
void vq_conv_gemm_lds(const InT* __restrict__ in, const bf16_t* __restrict__ w,
                      const float* __restrict__ bias, void* __restrict__ out) {
    __shared__ __align__(16) bf16_t bsh[16][40];  // 32 k + 8 pad (80B rows)

    constexpr int HW   = HO * WO;
    constexpr int HWIN = HIN * WIN;

    const int tid  = threadIdx.x;
    const int lane = tid & 31;
    const int wv   = tid >> 5;
    const int col  = lane & 15;       // position index within tile (N dim)
    const int h    = lane >> 4;       // K-half selector

    // --- staging role: thread stages 4 contiguous k for one position ---
    const int spos = tid >> 3;        // 0..15
    const int skg  = tid & 7;         // k-group: k = skg*4 + i
    const int sp   = (int)blockIdx.x * 16 + spos;
    const int sn   = sp / HW;
    const int srem = sp % HW;
    const int soy  = srem / WO, sox = srem % WO;
    const InT* sInN = in + (long)sn * CIN * HWIN;
    const int siy0 = soy * STRIDE - PAD;
    const int six0 = sox * STRIDE - PAD;

    // --- compute role: wave's output tile ---
    const int p    = (int)blockIdx.x * 16 + col;
    const int n    = p / HW;
    const int rem  = p % HW;
    const int mBase = (blockIdx.y * 4 + wv) * (16 * MT);
    const bf16_t* wrow = w + (long)(mBase + col) * KPAD;

    v8f acc[MT];
#pragma unroll
    for (int mt = 0; mt < MT; ++mt) acc[mt] = v8f{0.f, 0.f, 0.f, 0.f, 0.f, 0.f, 0.f, 0.f};

    for (int k0 = 0; k0 < KPAD; k0 += 32) {
        // ---- cooperative stage of B tile ----
        {
            const int kloc = skg * 4;
            const int kg   = k0 + kloc;
            if constexpr (KK == 4) {
                const int cin  = kg >> 4;                     // constant across i
                const int cinc = (KPAD == KREAL) ? cin : (kg < KREAL ? cin : 0);
                const InT* base = sInN + (long)cinc * HWIN;
                const int r  = kg & 15;                       // 4-aligned: one ky row
                const int iy = siy0 + (r >> 2);
                const bool oky = (unsigned)iy < (unsigned)HIN;
                const int iyc = clampi(iy, 0, HIN - 1);
#pragma unroll
                for (int i = 0; i < 4; ++i) {
                    const int ix = six0 + i;
                    bool ok = oky && (unsigned)ix < (unsigned)WIN;
                    if constexpr (KPAD != KREAL) ok = ok && (kg + i < KREAL);
                    const int ixc = clampi(ix, 0, WIN - 1);
                    const float v = (float)base[iyc * WIN + ixc];
                    bsh[spos][kloc + i] = (bf16_t)(ok ? v : 0.0f);
                }
            } else {  // KK==1, stride 1, pad 0: k == cin
#pragma unroll
                for (int i = 0; i < 4; ++i) {
                    const float v = (float)sInN[(long)(kg + i) * HWIN + srem];
                    bsh[spos][kloc + i] = (bf16_t)v;
                }
            }
        }
        __syncthreads();

        // ---- read shared B fragment, multiply-accumulate MT Cout tiles ----
        const v8bf lo = *(const v8bf*)&bsh[col][h * 8];
        const v8bf hi = *(const v8bf*)&bsh[col][16 + h * 8];
        const v16bf b = __builtin_shufflevector(lo, hi, 0, 1, 2, 3, 4, 5, 6, 7,
                                                        8, 9, 10, 11, 12, 13, 14, 15);
#pragma unroll
        for (int mt = 0; mt < MT; ++mt) {
            v16bf a = load_a_frag(wrow + (long)mt * 16 * KPAD, k0, h);
            acc[mt] = __builtin_amdgcn_wmma_f32_16x16x32_bf16(false, a, false, b,
                                                              (short)0, acc[mt], false, false);
        }
        __syncthreads();
    }

    // ---- epilogue: grid covers COUT exactly, no M guard ----
#pragma unroll
    for (int mt = 0; mt < MT; ++mt) {
#pragma unroll
        for (int r = 0; r < 8; ++r) {
            const int m = mBase + mt * 16 + r + 8 * h;
            float v = acc[mt][r] + bias[m];
            const long oidx = ((long)n * COUT + m) * (long)HW + rem;
            if (ACT == 1) {
                v = v > 0.0f ? v : 0.0f;
                ((bf16_t*)out)[oidx] = (bf16_t)v;
            } else {
                ((float*)out)[oidx] = v;
            }
        }
    }
}

// ---------------------------------------------------------------------------
// LDS-staged ConvTranspose2d(k=4,s=2,p=1) parity-class GEMM. blockIdx.z=class.
// Same block structure as vq_conv_gemm_lds. K = Cin*4, decode is shifts.
// Grid covers COUT exactly -> no M guard.
// ---------------------------------------------------------------------------
template <int ACT, int MT, int CIN, int HIN, int WIN, int COUT, int COUTPAD>
__global__ __launch_bounds__(128)
void vq_convt_gemm_lds(const bf16_t* __restrict__ in, const bf16_t* __restrict__ wcls,
                       const float* __restrict__ bias, void* __restrict__ out) {
    __shared__ __align__(16) bf16_t bsh[16][40];

    constexpr int KPAD = CIN * 4;
    constexpr int HWI  = HIN * WIN;

    const int cls = blockIdx.z;
    const int py = cls >> 1, px = cls & 1;
    const int tid  = threadIdx.x;
    const int lane = tid & 31;
    const int wv   = tid >> 5;
    const int col  = lane & 15;
    const int h    = lane >> 4;

    // staging role
    const int spos = tid >> 3;
    const int skg  = tid & 7;
    const int sp   = (int)blockIdx.x * 16 + spos;
    const int sn   = sp / HWI;
    const int srem = sp % HWI;
    const int sgy  = srem / WIN, sgx = srem % WIN;
    const bf16_t* sInN = in + (long)sn * CIN * HWI;
    const int siyA = sgy + py;        // taps iyA (ty=0), iyA-1 (ty=1)
    const int sixA = sgx + px;

    // compute role
    const int p    = (int)blockIdx.x * 16 + col;
    const int n    = p / HWI;
    const int rem  = p % HWI;
    const int gy   = rem / WIN, gx = rem % WIN;
    const int mBase = (blockIdx.y * 4 + wv) * (16 * MT);
    const bf16_t* wrow = wcls + (long)cls * COUTPAD * KPAD + (long)(mBase + col) * KPAD;

    v8f acc[MT];
#pragma unroll
    for (int mt = 0; mt < MT; ++mt) acc[mt] = v8f{0.f, 0.f, 0.f, 0.f, 0.f, 0.f, 0.f, 0.f};

    for (int k0 = 0; k0 < KPAD; k0 += 32) {
        {   // stage: 4-aligned k group -> one cin, taps t = i
            const int kloc = skg * 4;
            const int kg   = k0 + kloc;
            const int cin  = kg >> 2;
            const bf16_t* base = sInN + (long)cin * HWI;
#pragma unroll
            for (int i = 0; i < 4; ++i) {
                const int ty = i >> 1, tx = i & 1;
                const int iy = siyA - ty;
                const int ix = sixA - tx;
                const bool ok = (unsigned)iy < (unsigned)HIN && (unsigned)ix < (unsigned)WIN;
                const int iyc = clampi(iy, 0, HIN - 1);
                const int ixc = clampi(ix, 0, WIN - 1);
                const float v = (float)base[iyc * WIN + ixc];
                bsh[spos][kloc + i] = (bf16_t)(ok ? v : 0.0f);
            }
        }
        __syncthreads();

        const v8bf lo = *(const v8bf*)&bsh[col][h * 8];
        const v8bf hi = *(const v8bf*)&bsh[col][16 + h * 8];
        const v16bf b = __builtin_shufflevector(lo, hi, 0, 1, 2, 3, 4, 5, 6, 7,
                                                        8, 9, 10, 11, 12, 13, 14, 15);
#pragma unroll
        for (int mt = 0; mt < MT; ++mt) {
            v16bf a = load_a_frag(wrow + (long)mt * 16 * KPAD, k0, h);
            acc[mt] = __builtin_amdgcn_wmma_f32_16x16x32_bf16(false, a, false, b,
                                                              (short)0, acc[mt], false, false);
        }
        __syncthreads();
    }

    constexpr int HOO = 2 * HIN, WOO = 2 * WIN;
    const int oy = 2 * gy + py, ox = 2 * gx + px;
#pragma unroll
    for (int mt = 0; mt < MT; ++mt) {
#pragma unroll
        for (int r = 0; r < 8; ++r) {
            const int m = mBase + mt * 16 + r + 8 * h;
            float v = acc[mt][r] + bias[m];
            const long oidx = ((long)n * COUT + m) * ((long)HOO * WOO) + oy * WOO + ox;
            if (ACT == 1) {
                v = v > 0.0f ? v : 0.0f;
                ((bf16_t*)out)[oidx] = (bf16_t)v;
            } else {
                v = 1.0f / (1.0f + __expf(-v));
                ((float*)out)[oidx] = v;
            }
        }
    }
}

// ---------------------------------------------------------------------------
// Per-wave convT kernel (no LDS sharing) for the narrow final layer
// (COUT=3 < COUTPAD=16): 4 waves cover 4 consecutive position tiles.
// Keeps the M guard.
// ---------------------------------------------------------------------------
template <int ACT, int MT, int CIN, int HIN, int WIN, int COUT, int COUTPAD>
__global__ __launch_bounds__(128)
void vq_convt_gemm(const bf16_t* __restrict__ in, const bf16_t* __restrict__ wcls,
                   const float* __restrict__ bias, void* __restrict__ out) {
    constexpr int KPAD = CIN * 4;
    constexpr int HWI  = HIN * WIN;

    const int cls = blockIdx.z;
    const int py = cls >> 1, px = cls & 1;
    const int lane = threadIdx.x & 31;
    const int wv   = threadIdx.x >> 5;
    const int col  = lane & 15;
    const int h    = lane >> 4;
    const int p    = ((int)blockIdx.x * 4 + wv) * 16 + col;
    const int n    = p / HWI;
    const int rem  = p % HWI;
    const int gy   = rem / WIN, gx = rem % WIN;
    const int mBase = blockIdx.y * (16 * MT);
    const bf16_t* wrow = wcls + (long)cls * COUTPAD * KPAD + (long)(mBase + col) * KPAD;
    const bf16_t* inN  = in + (long)n * CIN * HWI;

    v8f acc[MT];
#pragma unroll
    for (int mt = 0; mt < MT; ++mt) acc[mt] = v8f{0.f, 0.f, 0.f, 0.f, 0.f, 0.f, 0.f, 0.f};

    const int iyA = gy + py;
    const int ixA = gx + px;

    for (int k0 = 0; k0 < KPAD; k0 += 32) {
        v16bf b;
#pragma unroll
        for (int half = 0; half < 2; ++half) {
            const int kr = k0 + h * 8 + half * 16;
#pragma unroll
            for (int j = 0; j < 8; ++j) {
                const int k  = kr + j;
                const int ty = (k >> 1) & 1;
                const int tx = k & 1;
                const int iy = iyA - ty;
                const int ix = ixA - tx;
                const bool ok = (unsigned)iy < (unsigned)HIN && (unsigned)ix < (unsigned)WIN;
                const int iyc = clampi(iy, 0, HIN - 1);
                const int ixc = clampi(ix, 0, WIN - 1);
                const float v = (float)inN[(long)(k >> 2) * HWI + iyc * WIN + ixc];
                b[half * 8 + j] = (bf16_t)(ok ? v : 0.0f);
            }
        }
#pragma unroll
        for (int mt = 0; mt < MT; ++mt) {
            v16bf a = load_a_frag(wrow + (long)mt * 16 * KPAD, k0, h);
            acc[mt] = __builtin_amdgcn_wmma_f32_16x16x32_bf16(false, a, false, b,
                                                              (short)0, acc[mt], false, false);
        }
    }

    constexpr int HOO = 2 * HIN, WOO = 2 * WIN;
    const int oy = 2 * gy + py, ox = 2 * gx + px;
#pragma unroll
    for (int mt = 0; mt < MT; ++mt) {
#pragma unroll
        for (int r = 0; r < 8; ++r) {
            const int m = mBase + mt * 16 + r + 8 * h;
            if (m < COUT) {
                float v = acc[mt][r] + bias[m];
                const long oidx = ((long)n * COUT + m) * ((long)HOO * WOO) + oy * WOO + ox;
                if (ACT == 1) {
                    v = v > 0.0f ? v : 0.0f;
                    ((bf16_t*)out)[oidx] = (bf16_t)v;
                } else {
                    v = 1.0f / (1.0f + __expf(-v));
                    ((float*)out)[oidx] = v;
                }
            }
        }
    }
}

// ---------------------------------------------------------------------------
// Vector quantization: one wave per token (D=64: 2 dims per lane).
// ---------------------------------------------------------------------------
__global__ __launch_bounds__(256)
void vq_quantize(const float* __restrict__ ze, const float* __restrict__ emb,
                 bf16_t* __restrict__ zq, float* __restrict__ loss,
                 float scale) {
    constexpr int D = 64, HW = 1024, K = 512;
    const int wave = blockIdx.x * 8 + (threadIdx.x >> 5);
    const int lane = threadIdx.x & 31;
    const int n = wave >> 10, hw = wave & 1023;
    const float* zp = ze + (long)n * D * HW + hw;
    const float x0 = zp[(2 * lane) * HW];
    const float x1 = zp[(2 * lane + 1) * HW];
    float best = 3.4e38f;
    int bidx = 0;
    for (int k = 0; k < K; ++k) {
        const float* e = emb + k * D + 2 * lane;
        float d0 = x0 - e[0];
        float d1 = x1 - e[1];
        float d = d0 * d0 + d1 * d1;
#pragma unroll
        for (int off = 16; off > 0; off >>= 1) d += __shfl_xor(d, off, 32);
        if (d < best) { best = d; bidx = k; }
    }
    const float* eb = emb + bidx * D + 2 * lane;
    bf16_t* zqp = zq + (long)n * D * HW + hw;
    zqp[(2 * lane) * HW]     = (bf16_t)eb[0];
    zqp[(2 * lane + 1) * HW] = (bf16_t)eb[1];
    if (lane == 0) atomicAdd(loss, best * scale);
}

// ---------------------------------------------------------------------------
// Host orchestration
// ---------------------------------------------------------------------------
extern "C" void kernel_launch(void* const* d_in, const int* in_sizes, int n_in,
                              void* d_out, int out_size, void* d_ws, size_t ws_size,
                              hipStream_t stream) {
    (void)in_sizes; (void)n_in; (void)out_size; (void)ws_size;
    const float* x   = (const float*)d_in[0];
    const float* ew1 = (const float*)d_in[1];  const float* eb1 = (const float*)d_in[2];
    const float* ew2 = (const float*)d_in[3];  const float* eb2 = (const float*)d_in[4];
    const float* ew3 = (const float*)d_in[5];  const float* eb3 = (const float*)d_in[6];
    const float* pw  = (const float*)d_in[7];  const float* pb  = (const float*)d_in[8];
    const float* emb = (const float*)d_in[9];
    const float* dw1 = (const float*)d_in[10]; const float* db1 = (const float*)d_in[11];
    const float* dw2 = (const float*)d_in[12]; const float* db2 = (const float*)d_in[13];
    const float* dw3 = (const float*)d_in[14]; const float* db3 = (const float*)d_in[15];

    // ---- workspace layout (bytes, 256-aligned) ----
    char* ws = (char*)d_ws;
    size_t off = 0;
    auto alloc = [&](size_t bytes) { size_t r = off; off += (bytes + 255) & ~(size_t)255; return r; };
    size_t o_w1  = alloc((size_t)128 * 64   * 2);
    size_t o_w2  = alloc((size_t)256 * 2048 * 2);
    size_t o_w3  = alloc((size_t)512 * 4096 * 2);
    size_t o_wp  = alloc((size_t)64  * 512  * 2);
    size_t o_wt1 = alloc((size_t)4 * 256 * 256  * 2);
    size_t o_wt2 = alloc((size_t)4 * 128 * 1024 * 2);
    size_t o_wt3 = alloc((size_t)4 * 16  * 512  * 2);
    size_t o_bufA = alloc((size_t)32 * 128 * 128 * 128 * 2); // act1 / dec2 (bf16)
    size_t o_bufB = alloc((size_t)32 * 256 * 64  * 64  * 2); // act2 / dec1 (bf16)
    size_t o_act3 = alloc((size_t)32 * 512 * 32  * 32  * 2); // act3 (bf16)
    size_t o_ze   = alloc((size_t)32 * 64  * 32  * 32  * 4); // z_e (f32)
    size_t o_zq   = alloc((size_t)32 * 64  * 32  * 32  * 2); // z_q (bf16)

    bf16_t* w1  = (bf16_t*)(ws + o_w1);
    bf16_t* w2  = (bf16_t*)(ws + o_w2);
    bf16_t* w3  = (bf16_t*)(ws + o_w3);
    bf16_t* wp  = (bf16_t*)(ws + o_wp);
    bf16_t* wt1 = (bf16_t*)(ws + o_wt1);
    bf16_t* wt2 = (bf16_t*)(ws + o_wt2);
    bf16_t* wt3 = (bf16_t*)(ws + o_wt3);
    bf16_t* bufA = (bf16_t*)(ws + o_bufA);
    bf16_t* bufB = (bf16_t*)(ws + o_bufB);
    bf16_t* act3 = (bf16_t*)(ws + o_act3);
    float*  ze   = (float*) (ws + o_ze);
    bf16_t* zq   = (bf16_t*)(ws + o_zq);

    float* y_out    = (float*)d_out;              // 32*3*256*256
    float* loss_out = y_out + (size_t)32 * 3 * 256 * 256;

    auto grid1d = [](long n) { return (unsigned)((n + 255) / 256); };

    // ---- weight repack ----
    vq_pack_conv_w<<<grid1d((long)128 * 64), 256, 0, stream>>>(ew1, w1, 128, 128, 48, 64);
    vq_pack_conv_w<<<grid1d((long)256 * 2048), 256, 0, stream>>>(ew2, w2, 256, 256, 2048, 2048);
    vq_pack_conv_w<<<grid1d((long)512 * 4096), 256, 0, stream>>>(ew3, w3, 512, 512, 4096, 4096);
    vq_pack_conv_w<<<grid1d((long)64 * 512), 256, 0, stream>>>(pw, wp, 64, 64, 512, 512);
    vq_pack_convt_w<<<grid1d(4L * 256 * 256), 256, 0, stream>>>(dw1, wt1, 64, 256, 256);
    vq_pack_convt_w<<<grid1d(4L * 128 * 1024), 256, 0, stream>>>(dw2, wt2, 256, 128, 128);
    vq_pack_convt_w<<<grid1d(4L * 16 * 512), 256, 0, stream>>>(dw3, wt3, 128, 3, 16);
    vq_zero_scalar<<<1, 32, 0, stream>>>(loss_out);

    // ---- encoder ----  (grid.x = P/16; grid.y * 4waves * MT * 16 = COUT)
    // conv1: 3->128, 256x256 -> 128x128, K=48 pad 64, MT=2 (4*2*16=128)
    vq_conv_gemm_lds<float, 1, 2, 4, 2, 1, 48, 64, 3, 256, 256, 128, 128, 128>
        <<<dim3(32768, 1, 1), 128, 0, stream>>>(x, w1, eb1, bufA);
    // conv2: 128->256, 128x128 -> 64x64, K=2048, MT=4 (4*4*16=256)
    vq_conv_gemm_lds<bf16_t, 1, 4, 4, 2, 1, 2048, 2048, 128, 128, 128, 256, 64, 64>
        <<<dim3(8192, 1, 1), 128, 0, stream>>>(bufA, w2, eb2, bufB);
    // conv3: 256->512, 64x64 -> 32x32, K=4096, MT=4, gy=2 (2*256=512)
    vq_conv_gemm_lds<bf16_t, 1, 4, 4, 2, 1, 4096, 4096, 256, 64, 64, 512, 32, 32>
        <<<dim3(2048, 2, 1), 128, 0, stream>>>(bufB, w3, eb3, act3);
    // pre-VQ 1x1: 512->64, linear, f32 out, MT=1 (4*1*16=64)
    vq_conv_gemm_lds<bf16_t, 0, 1, 1, 1, 0, 512, 512, 512, 32, 32, 64, 32, 32>
        <<<dim3(2048, 1, 1), 128, 0, stream>>>(act3, wp, pb, ze);

    // ---- vector quantization ----
    const float scale = 2.0f / (32768.0f * 64.0f);
    vq_quantize<<<4096, 256, 0, stream>>>(ze, emb, zq, loss_out, scale);

    // ---- decoder (4 parity classes per layer) ----
    // convT1: 64->256, 32x32 -> 64x64, K=256, MT=4 (4*4*16=256)
    vq_convt_gemm_lds<1, 4, 64, 32, 32, 256, 256>
        <<<dim3(2048, 1, 4), 128, 0, stream>>>(zq, wt1, db1, bufB);
    // convT2: 256->128, 64x64 -> 128x128, K=1024, MT=2 (4*2*16=128)
    vq_convt_gemm_lds<1, 2, 256, 64, 64, 128, 128>
        <<<dim3(8192, 1, 4), 128, 0, stream>>>(bufB, wt2, db2, bufA);
    // convT3: 128->3 (pad 16), 128x128 -> 256x256, K=512, sigmoid -> f32 d_out
    // narrow M: per-wave kernel, 4 waves on 4 position tiles
    vq_convt_gemm<2, 1, 128, 128, 128, 3, 16>
        <<<dim3(8192, 1, 4), 128, 0, stream>>>(bufA, wt3, db3, y_out);
}
